// ByteSpectralEmbedding_27212912788370
// MI455X (gfx1250) — compile-verified, matrix operand order checked by reading, not run on hardware
//
#include <hip/hip_runtime.h>
#include <hip/hip_fp16.h>
#include <math.h>

typedef __attribute__((ext_vector_type(16))) _Float16 v16h;
typedef __attribute__((ext_vector_type(8)))  _Float16 v8h;
typedef __attribute__((ext_vector_type(8)))  float    v8f;

#define B_SZ   32
#define T_LEN  4096
#define D_M    256
#define N1     512      // 2*D
#define K1     128      // sin-feature count (= k)
#define TWO_PI 6.28318530717958647692f

// LDS layout of the fused kernel (dynamic shared segment starts at offset 0)
#define LDS_HST   0         // 8 waves * 16 rows * 512 f16  = 128 KB
#define LDS_W1    131072    // packed w1 B-fragments (f16)  = 128 KB
#define LDS_W2    262144    // w2 K-slice double buffer 2*16KB = 32 KB
#define LDS_TOTAL 294912    // 288 KB  (<= 320 KB per WGP)

// K index within a 32-wide K-tile for element e (0..15) of a 16-bit A/B fragment.
// Lanes 0-15: e0..7 -> K 0..7,  e8..15 -> K 16..23
// Lanes16-31: e0..7 -> K 8..15, e8..15 -> K 24..31
__host__ __device__ __forceinline__ int kmap(int e, int hi) {
    return (e < 8 ? e : e + 8) + (hi ? 8 : 0);
}

// ---- CDNA5 async global->LDS copy (ASYNCcnt-tracked), GVS addressing -------
__device__ __forceinline__ void async_copy_b128(unsigned lds_off, const void* gbase,
                                                unsigned goff) {
    asm volatile("global_load_async_to_lds_b128 %0, %1, %2 offset:0"
                 :: "v"(lds_off), "v"(goff), "s"(gbase)
                 : "memory");
}
__device__ __forceinline__ void wait_async0() {
    asm volatile("s_wait_asynccnt 0" ::: "memory");
}

// ---------------------------------------------------------------------------
// Kernel 1: direct DFT, 128 bins per batch. One block per (b, f).
// ---------------------------------------------------------------------------
__global__ __launch_bounds__(256) void dft_kernel(
    const int* __restrict__ byte_ids, const float* __restrict__ freq_bands,
    float* __restrict__ magfb, float* __restrict__ ph0)
{
    const int b = blockIdx.x >> 7;
    const int f = blockIdx.x & 127;
    const int tid = threadIdx.x;
    __shared__ float sre[256];
    __shared__ float sim[256];

    const float w = -TWO_PI / (float)T_LEN;
    float re = 0.f, im = 0.f;
    for (int t = tid; t < T_LEN; t += 256) {
        float s = (float)byte_ids[b * T_LEN + t] * (1.0f / 127.5f) - 1.0f;
        float ang = w * (float)((t * f) & (T_LEN - 1));   // exact mod-T reduction
        float sn, cs;
        __sincosf(ang, &sn, &cs);
        re += s * cs;
        im += s * sn;
    }
    sre[tid] = re; sim[tid] = im;
    __syncthreads();
    for (int s = 128; s > 0; s >>= 1) {
        if (tid < s) { sre[tid] += sre[tid + s]; sim[tid] += sim[tid + s]; }
        __syncthreads();
    }
    if (tid == 0) {
        float R = sre[0], I = sim[0];
        magfb[b * K1 + f] = sqrtf(R * R + I * I) * freq_bands[f];
        ph0  [b * K1 + f] = atan2f(I, R);
    }
}

// ---------------------------------------------------------------------------
// Kernel 2: cvec[b,j] = b1[j] + sum_f magfb[b,f] * w1[f, j]   (t-invariant)
// ---------------------------------------------------------------------------
__global__ __launch_bounds__(256) void cvec_kernel(
    const float* __restrict__ magfb, const float* __restrict__ w1,
    const float* __restrict__ b1, float* __restrict__ cvec)
{
    int idx = blockIdx.x * 256 + threadIdx.x;
    int b = idx >> 9, j = idx & (N1 - 1);
    float s = b1[j];
    #pragma unroll 4
    for (int f = 0; f < K1; ++f)
        s += magfb[b * K1 + f] * w1[f * N1 + j];
    cvec[idx] = s;
}

// ---------------------------------------------------------------------------
// Kernel 3a/3b: pack weights into f16 WMMA B-fragment layout
// ---------------------------------------------------------------------------
__global__ __launch_bounds__(256) void pack_w1_kernel(
    const float* __restrict__ w1, _Float16* __restrict__ w1p)
{
    int idx = blockIdx.x * 256 + threadIdx.x;     // [kt<4][ntg<32][lane<32][e<16]
    int e = idx & 15, lane = (idx >> 4) & 31, ntg = (idx >> 9) & 31, kt = idx >> 14;
    int k = kt * 32 + kmap(e, lane >= 16);
    int n = ntg * 16 + (lane & 15);
    w1p[idx] = (_Float16)w1[(K1 + k) * N1 + n];
}

__global__ __launch_bounds__(256) void pack_w2_kernel(
    const float* __restrict__ w2, _Float16* __restrict__ w2p)
{
    int idx = blockIdx.x * 256 + threadIdx.x;     // [kt<16][nt<16][lane<32][e<16]
    int e = idx & 15, lane = (idx >> 4) & 31, nt = (idx >> 9) & 15, kt = idx >> 13;
    int k = kt * 32 + kmap(e, lane >= 16);
    int n = nt * 16 + (lane & 15);
    w2p[idx] = (_Float16)w2[k * D_M + n];
}

// ---------------------------------------------------------------------------
// Main fused kernel.
// Grid: 1024 WGs x 256 threads (8 wave32s). Each wave owns 16 rows (t's).
// ---------------------------------------------------------------------------
__global__ __launch_bounds__(256) void fused_main_kernel(
    const float* __restrict__ ph0,      // [B,128]
    const float* __restrict__ cvec,     // [B,512]
    const _Float16* __restrict__ w1p,   // [4][32][32][16] packed f16
    const _Float16* __restrict__ w2p,   // [16][16][32][16] packed f16
    const float* __restrict__ gamma,    // [512]
    const float* __restrict__ beta,     // [512]
    const float* __restrict__ b2,       // [256]
    float* __restrict__ out)            // [B,T,256]
{
    extern __shared__ char smem[];
    _Float16* hst   = (_Float16*)(smem + LDS_HST);
    const _Float16* w1lds = (const _Float16*)(smem + LDS_W1);
    const _Float16* w2lds = (const _Float16*)(smem + LDS_W2);

    const int tid  = threadIdx.x;
    const int wave = tid >> 5;
    const int lane = tid & 31;
    const int m    = lane & 15;
    const int hi   = (lane >= 16) ? 1 : 0;

    const int wg = blockIdx.x;
    const int b  = wg >> 5;
    const int t0 = ((wg & 31) * 8 + wave) * 16;
    const int t  = t0 + m;

    // ---- Kick off async staging: full w1 pack (128KB) + w2 K-slice 0 ------
    #pragma unroll 4
    for (int i = tid; i < 8192; i += 256)
        async_copy_b128(LDS_W1 + i * 16, w1p, (unsigned)(i * 16));
    #pragma unroll
    for (int i = tid; i < 1024; i += 256)
        async_copy_b128(LDS_W2 + i * 16, w2p, (unsigned)(i * 16));

    // ---- Build GEMM1 A fragments while the DMA flies ----------------------
    v16h a1[4];
    const float wph = TWO_PI / (float)T_LEN;
    #pragma unroll
    for (int kt = 0; kt < 4; ++kt) {
        #pragma unroll
        for (int e = 0; e < 16; ++e) {
            int k = kt * 32 + kmap(e, hi);
            float phase = ph0[b * K1 + k] + wph * (float)((t * k) & (T_LEN - 1));
            a1[kt][e] = (_Float16)__sinf(phase);
        }
    }
    wait_async0();
    __syncthreads();                      // w1 fragments (and w2 slice 0) ready

    // ---- GEMM1 (two N-halves of 256) with fused stats accumulation --------
    float s8[8]  = {};
    float ss8[8] = {};
    #pragma unroll
    for (int half = 0; half < 2; ++half) {
        v8f acc[16] = {};
        // software-pipelined B-fragment stream (distance-1 prefetch from LDS)
        v16h bcur = *(const v16h*)(w1lds + (((half * 16) * 32 + lane) << 4));
        #pragma unroll
        for (int kt = 0; kt < 4; ++kt) {
            #pragma unroll
            for (int nt = 0; nt < 16; ++nt) {
                v16h bnxt = bcur;
                int nxt = kt * 16 + nt + 1;            // flattened (kt,nt) + 1
                if (nxt < 64) {
                    int nkt = nxt >> 4, nnt = nxt & 15;
                    bnxt = *(const v16h*)(w1lds +
                            (((nkt * 32 + half * 16 + nnt) * 32 + lane) << 4));
                }
                acc[nt] = __builtin_amdgcn_wmma_f32_16x16x32_f16(
                    false, a1[kt], false, bcur, (short)0, acc[nt], false, false);
                bcur = bnxt;
            }
        }
        #pragma unroll
        for (int nt = 0; nt < 16; ++nt) {
            int n = half * 256 + nt * 16 + m;
            float cv = cvec[b * N1 + n];
            #pragma unroll
            for (int r = 0; r < 8; ++r) {
                float v = acc[nt][r] + cv;                 // pre-LN h (f32)
                s8[r]  += v;
                ss8[r] += v * v;
                hst[(((wave * 16) + r + (hi ? 8 : 0)) << 9) + n] = (_Float16)v;
            }
        }
    }

    // ---- LN stats: butterfly within 16-lane halves (rows r / r+8) ---------
    float mu8[8], rs8[8];
    #pragma unroll
    for (int r = 0; r < 8; ++r) {
        float s = s8[r], ss = ss8[r];
        #pragma unroll
        for (int d = 1; d < 16; d <<= 1) {
            s  += __shfl_xor(s, d);
            ss += __shfl_xor(ss, d);
        }
        float mu  = s * (1.0f / (float)N1);
        mu8[r] = mu;
        rs8[r] = rsqrtf(ss * (1.0f / (float)N1) - mu * mu + 1e-5f);
    }

    // ---- LN + exact GELU rewrite (each lane rewrites its own elements) ----
    #pragma unroll 4
    for (int j = 0; j < 32; ++j) {
        int n = j * 16 + m;
        float g = gamma[n], bb = beta[n];
        #pragma unroll
        for (int r = 0; r < 8; ++r) {
            unsigned idx = (((wave * 16) + r + (hi ? 8 : 0)) << 9) + n;
            float x = (float)hst[idx];
            x = (x - mu8[r]) * rs8[r] * g + bb;
            float ge = 0.5f * x * (1.0f + erff(x * 0.70710678118654752f));
            hst[idx] = (_Float16)ge;
        }
    }

    // ---- GEMM2: stream w2 K-slices (double-buffered async), ds+wmma -------
    v8f acc2[16] = {};
    const unsigned rowoff = ((unsigned)(wave * 16 + m)) << 9;   // my A-row base
    for (int kt = 0; kt < 16; ++kt) {
        wait_async0();
        __syncthreads();                  // slice kt resident; buf (kt+1)&1 free
        if (kt < 15) {
            unsigned dst = LDS_W2 + (((kt + 1) & 1) ? 16384u : 0u);
            unsigned src = (unsigned)((kt + 1) * 16384);
            #pragma unroll
            for (int i = tid; i < 1024; i += 256)
                async_copy_b128(dst + i * 16, w2p, src + i * 16);
        }
        // A fragment: two contiguous 8-half groups from my staged (LN'd) row
        int kb = kt * 32 + (hi ? 8 : 0);
        v8h g0 = *(const v8h*)(hst + rowoff + kb);
        v8h g1 = *(const v8h*)(hst + rowoff + kb + 16);
        v16h a2;
        #pragma unroll
        for (int i = 0; i < 8; ++i) { a2[i] = g0[i]; a2[i + 8] = g1[i]; }

        const _Float16* wb = w2lds + ((kt & 1) ? 8192 : 0);
        // software-pipelined B-fragment stream (distance-1 prefetch from LDS)
        v16h bcur = *(const v16h*)(wb + (lane << 4));
        #pragma unroll
        for (int nt = 0; nt < 16; ++nt) {
            v16h bnxt = bcur;
            if (nt < 15)
                bnxt = *(const v16h*)(wb + (((nt + 1) * 32 + lane) << 4));
            acc2[nt] = __builtin_amdgcn_wmma_f32_16x16x32_f16(
                false, a2, false, bcur, (short)0, acc2[nt], false, false);
            bcur = bnxt;
        }
    }

    // ---- Epilogue: + b2, store f32 ----------------------------------------
    #pragma unroll
    for (int nt = 0; nt < 16; ++nt) {
        int n = nt * 16 + m;
        float bias = b2[n];
        #pragma unroll
        for (int r = 0; r < 8; ++r) {
            int row = r + (hi ? 8 : 0);
            out[((size_t)(b * T_LEN + t0 + row)) * D_M + n] = acc2[nt][r] + bias;
        }
    }
}

// ---------------------------------------------------------------------------
// Launcher
// ---------------------------------------------------------------------------
extern "C" void kernel_launch(void* const* d_in, const int* in_sizes, int n_in,
                              void* d_out, int out_size, void* d_ws, size_t ws_size,
                              hipStream_t stream) {
    (void)in_sizes; (void)n_in; (void)out_size; (void)ws_size;
    const int*   byte_ids = (const int*)  d_in[0];
    const float* freq     = (const float*)d_in[1];
    const float* w1       = (const float*)d_in[2];
    const float* b1       = (const float*)d_in[3];
    const float* gamma    = (const float*)d_in[4];
    const float* beta     = (const float*)d_in[5];
    const float* w2       = (const float*)d_in[6];
    const float* b2       = (const float*)d_in[7];
    float* out = (float*)d_out;

    char* ws = (char*)d_ws;
    float*    magfb = (float*)   (ws + 0);        //  16 KB
    float*    ph0   = (float*)   (ws + 16384);    //  16 KB
    float*    cvec  = (float*)   (ws + 32768);    //  64 KB
    _Float16* w1p   = (_Float16*)(ws + 98304);    // 128 KB
    _Float16* w2p   = (_Float16*)(ws + 229376);   // 256 KB   (total 480 KB)

    dft_kernel    <<<B_SZ * K1,          256, 0, stream>>>(byte_ids, freq, magfb, ph0);
    cvec_kernel   <<<(B_SZ * N1) / 256,  256, 0, stream>>>(magfb, w1, b1, cvec);
    pack_w1_kernel<<<65536 / 256,        256, 0, stream>>>(w1, w1p);
    pack_w2_kernel<<<131072 / 256,       256, 0, stream>>>(w2, w2p);

    fused_main_kernel<<<(B_SZ * T_LEN) / 128, 256, LDS_TOTAL, stream>>>(
        ph0, cvec, w1p, w2p, gamma, beta, b2, out);
}
